// Representation_32074815766664
// MI455X (gfx1250) — matrix operand order repaired and checked by reading, compiled, not run on hardware
//
#include <hip/hip_runtime.h>
#include <stdint.h>

// ---------------- types ----------------
typedef __bf16 bf16_t;
typedef __attribute__((ext_vector_type(8)))  float  v8f;
typedef __attribute__((ext_vector_type(16))) __bf16 v16bf;
typedef __attribute__((ext_vector_type(8)))  __bf16 v8bf;

// ---------------- problem constants ----------------
#define Lq    20
#define Nq    512
#define Cq    10
#define OPLq  13
#define HIDq  512
#define MIDq  256
#define GQ    2048      // 4*HID gate width
#define K1q   544       // 32 (padded OPL) + 512 ; 17 k-tiles
#define KT1   17
#define FEAT  640       // 635 padded to mult of 32
#define XF    635       // 15 + 108 + 256 + 256
#define K2q   1152      // 640 + 512 ; 36 k-tiles
#define KT2   36
#define ROWS1 10240     // L*N
#define EPSq  1e-5f

// element strides (in bf16 elements) inside a k-tile-major weight slab
#define GSTRIDE (HIDq * 32)   // one gate block within a tile: 16384 elems
#define JSTRIDE (16 * 32)     // one 16-col group within a gate: 512 elems

// ---------------- device helpers ----------------
__device__ __forceinline__ v16bf bf_cat(v8bf lo, v8bf hi) {
  v16bf r;
#pragma unroll
  for (int i = 0; i < 8; ++i) { r[i] = lo[i]; r[8 + i] = hi[i]; }
  return r;
}

// A fragment chunks: lane already folded into p; {p, p+16}
__device__ __forceinline__ v16bf ldA(const bf16_t* p) {
  return bf_cat(*(const v8bf*)p, *(const v8bf*)(p + 16));
}
// B fragment chunks: lane already folded into p; {p, p+16}
__device__ __forceinline__ v16bf ldB(const bf16_t* p) {
  return bf_cat(*(const v8bf*)p, *(const v8bf*)(p + 16));
}

__device__ __forceinline__ v8f wmma_bf16(v16bf a, v16bf b, v8f c) {
  return __builtin_amdgcn_wmma_f32_16x16x32_bf16(false, a, false, b,
                                                 (short)0, c, false, false);
}

__device__ __forceinline__ float sigmf(float x) {
  return 1.0f / (1.0f + __expf(-x));
}

// ---------------- prep kernels (k-tile-major weight packing) ----------------
__global__ void prep_weights_kernel(const float* w_ih1, const float* w_hh1,
                                    const float* w_ih2, const float* w_hh2,
                                    const float* sample_w, const float* cond_w,
                                    bf16_t* Wcat1, bf16_t* Wcat2,
                                    bf16_t* swpad, bf16_t* cwbf,
                                    float* stats /* 1024 floats zeroed */) {
  int stride = gridDim.x * blockDim.x;
  int t0 = blockIdx.x * blockDim.x + threadIdx.x;
  // Wcat1: [17][2048][32] <- [W_ih1 pad32 | W_hh1]
  for (int i = t0; i < GQ * K1q; i += stride) {
    int n = i / K1q, k = i % K1q;
    float v = (k < 32) ? (k < OPLq ? w_ih1[n * OPLq + k] : 0.f)
                       : w_hh1[n * HIDq + (k - 32)];
    Wcat1[((size_t)(k >> 5) * GQ + n) * 32 + (k & 31)] = (bf16_t)v;
  }
  // Wcat2: [36][2048][32] <- [W_ih2 pad640 | W_hh2]
  for (int i = t0; i < GQ * K2q; i += stride) {
    int n = i / K2q, k = i % K2q;
    float v = (k < FEAT) ? (k < XF ? w_ih2[n * XF + k] : 0.f)
                         : w_hh2[n * HIDq + (k - FEAT)];
    Wcat2[((size_t)(k >> 5) * GQ + n) * 32 + (k & 31)] = (bf16_t)v;
  }
  // swpad: [32][256][32]
  for (int i = t0; i < MIDq * 1024; i += stride) {
    int c = i >> 10, k = i & 1023;
    float v = (k < 1000) ? sample_w[c * 1000 + k] : 0.f;
    swpad[((size_t)(k >> 5) * MIDq + c) * 32 + (k & 31)] = (bf16_t)v;
  }
  // cwbf: [16][256][32]
  for (int i = t0; i < MIDq * HIDq; i += stride) {
    int c = i / HIDq, k = i % HIDq;
    cwbf[((size_t)(k >> 5) * MIDq + c) * 32 + (k & 31)] = (bf16_t)cond_w[i];
  }
  for (int i = t0; i < 1024; i += stride) stats[i] = 0.f;
}

__global__ void prep_acts_kernel(const float* samples, const float* operators,
                                 const float* extra, bf16_t* sampad,
                                 bf16_t* feats) {
  int stride = gridDim.x * blockDim.x;
  int t0 = blockIdx.x * blockDim.x + threadIdx.x;
  for (int i = t0; i < ROWS1 * 1024; i += stride) {
    int r = i >> 10, k = i & 1023;
    sampad[i] = (bf16_t)(k < 1000 ? samples[(size_t)r * 1000 + k] : 0.f);
  }
  for (int i = t0; i < ROWS1 * FEAT; i += stride) {
    int r = i / FEAT, c = i % FEAT;
    if (c < 15)        feats[i] = (bf16_t)operators[r * 15 + c];
    else if (c < 123)  feats[i] = (bf16_t)extra[r * 108 + (c - 15)];
    else if (c >= XF)  feats[i] = (bf16_t)0.f;
  }
}

// ---------------- LSTM1: 512 threads, 32-row tile, pipelined B ----------------
__global__ __launch_bounds__(512) void lstm1_kernel(const float* cond1,
                                                    const bf16_t* Wcat1,
                                                    const float* b_ih,
                                                    const float* b_hh,
                                                    bf16_t* h_out) {
  __shared__ __align__(16) bf16_t sh_h[32 * HIDq];
  __shared__ __align__(16) bf16_t sh_x[32 * 32];
  const int row0 = blockIdx.x * 32;
  const int tid = threadIdx.x;
  const int wave = tid >> 5, lane = tid & 31;
  const int mtile = wave >> 3;          // 0..1 : which 16-row tile
  const int wq = wave & 7;              // 0..7 : 64-col hidden slice
  const int mlane = lane & 15, hh = lane >> 4;

  // single lane-dependent base for the whole B stream
  const bf16_t* wlane = Wcat1 + (size_t)(wq * 64 + mlane) * 32 + hh * 8;
  // lane-dependent A bases (LDS)
  const bf16_t* hlane = sh_h + (size_t)(mtile * 16 + mlane) * HIDq + hh * 8;
  const bf16_t* xlane = sh_x + (size_t)(mtile * 16 + mlane) * 32 + hh * 8;

  for (int i = tid; i < 32 * HIDq; i += 512) sh_h[i] = (bf16_t)0.f;
  v8f cfrag[4] = {};

  // hoist gate biases out of the timestep loop
  float Bi[4], Bff[4], Bg[4], Bo[4];
#pragma unroll
  for (int jt = 0; jt < 4; ++jt) {
    int j = wq * 64 + jt * 16 + mlane;
    Bi[jt]  = b_ih[j] + b_hh[j];
    Bff[jt] = b_ih[HIDq + j] + b_hh[HIDq + j];
    Bg[jt]  = b_ih[2 * HIDq + j] + b_hh[2 * HIDq + j];
    Bo[jt]  = b_ih[3 * HIDq + j] + b_hh[3 * HIDq + j];
  }

  for (int t = 0; t < Cq; ++t) {
    for (int i = tid; i < 32 * 32; i += 512) {
      int r = i >> 5, cc = i & 31;
      sh_x[i] = (cc < OPLq)
                    ? (bf16_t)cond1[((size_t)(row0 + r) * Cq + t) * OPLq + cc]
                    : (bf16_t)0.f;
    }
    __syncthreads();

    v8f acc[16] = {};
    const bf16_t* wk = wlane;           // advances by one tile per kt
    for (int kt = 0; kt < KT1; ++kt) {
      v16bf A = (kt == 0) ? ldA(xlane) : ldA(hlane + (kt - 1) * 32);
      // double-buffered batches of 4 B fragments (one per gate)
      v16bf B0 = ldB(wk);
      v16bf B1 = ldB(wk + GSTRIDE);
      v16bf B2 = ldB(wk + 2 * GSTRIDE);
      v16bf B3 = ldB(wk + 3 * GSTRIDE);
#pragma unroll
      for (int jt = 0; jt < 4; ++jt) {
        v16bf N0, N1, N2, N3;
        if (jt < 3) {
          const bf16_t* q = wk + (jt + 1) * JSTRIDE;
          N0 = ldB(q);
          N1 = ldB(q + GSTRIDE);
          N2 = ldB(q + 2 * GSTRIDE);
          N3 = ldB(q + 3 * GSTRIDE);
        }
        acc[jt * 4 + 0] = wmma_bf16(A, B0, acc[jt * 4 + 0]);
        acc[jt * 4 + 1] = wmma_bf16(A, B1, acc[jt * 4 + 1]);
        acc[jt * 4 + 2] = wmma_bf16(A, B2, acc[jt * 4 + 2]);
        acc[jt * 4 + 3] = wmma_bf16(A, B3, acc[jt * 4 + 3]);
        if (jt < 3) { B0 = N0; B1 = N1; B2 = N2; B3 = N3; }
      }
      wk += (size_t)GQ * 32;
    }
    __syncthreads();  // all reads of sh_h done before rewrite

#pragma unroll
    for (int jt = 0; jt < 4; ++jt) {
      int j = wq * 64 + jt * 16 + mlane;
#pragma unroll
      for (int e = 0; e < 8; ++e) {
        float iv = sigmf(acc[jt * 4 + 0][e] + Bi[jt]);
        float fv = sigmf(acc[jt * 4 + 1][e] + Bff[jt]);
        float gv = tanhf(acc[jt * 4 + 2][e] + Bg[jt]);
        float ov = sigmf(acc[jt * 4 + 3][e] + Bo[jt]);
        float cn = fv * cfrag[jt][e] + iv * gv;
        cfrag[jt][e] = cn;
        int r = e + 8 * hh;
        sh_h[(mtile * 16 + r) * HIDq + j] = (bf16_t)(ov * tanhf(cn));
      }
    }
    __syncthreads();
  }
  for (int i = tid; i < 32 * HIDq; i += 512)
    h_out[(size_t)row0 * HIDq + i] = sh_h[i];
}

// ---------------- cond head: relu(h1 @ cond_w.T + b) + BN stats ----------------
__global__ __launch_bounds__(256) void cond_gemm_kernel(
    const bf16_t* h1, const bf16_t* cw, const float* bias, float* pre,
    float* sums, float* sqs) {
  const int row0 = blockIdx.x * 16;
  const int tid = threadIdx.x;
  const int wave = tid >> 5, lane = tid & 31;
  const int mlane = lane & 15, hh = lane >> 4;
  const bf16_t* alane = h1 + (size_t)(row0 + mlane) * HIDq + hh * 8;
  const bf16_t* wk = cw + (size_t)(wave * 32 + mlane) * 32 + hh * 8;
  v8f acc[2] = {};
  for (int kt = 0; kt < HIDq / 32; ++kt) {
    v16bf A = ldA(alane);
    v16bf B0 = ldB(wk);
    v16bf B1 = ldB(wk + JSTRIDE);
    acc[0] = wmma_bf16(A, B0, acc[0]);
    acc[1] = wmma_bf16(A, B1, acc[1]);
    alane += 32;
    wk += (size_t)MIDq * 32;
  }
#pragma unroll
  for (int nt = 0; nt < 2; ++nt) {
    int col = wave * 32 + nt * 16 + mlane;
    float b = bias[col];
    float ls = 0.f, lq = 0.f;
#pragma unroll
    for (int e = 0; e < 8; ++e) {
      float v = fmaxf(acc[nt][e] + b, 0.f);
      int r = e + 8 * hh;
      pre[(size_t)(row0 + r) * MIDq + col] = v;
      ls += v; lq += v * v;
    }
    atomicAdd(&sums[col], ls);
    atomicAdd(&sqs[col], lq);
  }
}

__global__ void bn1_apply_kernel(const float* pre, const float* sums,
                                 const float* sqs, const float* g,
                                 const float* b, bf16_t* feats) {
  int stride = gridDim.x * blockDim.x;
  for (int i = blockIdx.x * blockDim.x + threadIdx.x; i < ROWS1 * MIDq;
       i += stride) {
    int r = i >> 8, c = i & 255;
    float mean = sums[c] * (1.0f / ROWS1);
    float var = sqs[c] * (1.0f / ROWS1) - mean * mean;
    float y = (pre[i] - mean) * rsqrtf(var + EPSq) * g[c] + b[c];
    feats[(size_t)r * FEAT + 123 + c] = (bf16_t)y;
  }
}

// ---------------- samples MLP: relu(s @ sw.T + b) * mask -> feats ----------------
__global__ __launch_bounds__(256) void samp_gemm_kernel(const bf16_t* sampad,
                                                        const bf16_t* sw,
                                                        const float* bias,
                                                        const float* cmask,
                                                        bf16_t* feats) {
  const int row0 = blockIdx.x * 16;
  const int tid = threadIdx.x;
  const int wave = tid >> 5, lane = tid & 31;
  const int mlane = lane & 15, hh = lane >> 4;
  const bf16_t* alane = sampad + (size_t)(row0 + mlane) * 1024 + hh * 8;
  const bf16_t* wk = sw + (size_t)(wave * 32 + mlane) * 32 + hh * 8;
  v8f acc[2] = {};
  for (int kt = 0; kt < 1024 / 32; ++kt) {
    v16bf A = ldA(alane);
    v16bf B0 = ldB(wk);
    v16bf B1 = ldB(wk + JSTRIDE);
    acc[0] = wmma_bf16(A, B0, acc[0]);
    acc[1] = wmma_bf16(A, B1, acc[1]);
    alane += 32;
    wk += (size_t)MIDq * 32;
  }
#pragma unroll
  for (int nt = 0; nt < 2; ++nt) {
    int col = wave * 32 + nt * 16 + mlane;
    float b = bias[col];
#pragma unroll
    for (int e = 0; e < 8; ++e) {
      int r = e + 8 * hh;
      float v = fmaxf(acc[nt][e] + b, 0.f) * cmask[row0 + r];
      feats[(size_t)(row0 + r) * FEAT + 379 + col] = (bf16_t)v;
    }
  }
}

// ---------------- tree-LSTM2: one launch per level ----------------
__global__ __launch_bounds__(256) void lstm2_level_kernel(
    const bf16_t* feats, const bf16_t* Wcat2, const float* b_ih,
    const float* b_hh, const int* mapping, const float* hsrc,
    const float* csrc, float* hdst, float* cdst, int level, int first) {
  __shared__ __align__(16) bf16_t sh_h[16 * HIDq];
  __shared__ float sh_c[16 * HIDq];
  const int row0 = blockIdx.x * 16;
  const int tid = threadIdx.x;
  const int wave = tid >> 5, lane = tid & 31;
  const int mlane = lane & 15, hh = lane >> 4;

  // gather-average children states (index 0 == zero pad)
  for (int i = tid; i < 16 * HIDq; i += 256) {
    int r = i >> 9, cc = i & (HIDq - 1);
    float hv = 0.f, cv = 0.f;
    if (!first) {
      int rg = row0 + r;
      int m0 = mapping[level * 2 * Nq + rg];
      int m1 = mapping[level * 2 * Nq + Nq + rg];
      float h0 = 0.f, c0 = 0.f, h1 = 0.f, c1 = 0.f;
      if (m0 > 0) { h0 = hsrc[(size_t)(m0 - 1) * HIDq + cc];
                    c0 = csrc[(size_t)(m0 - 1) * HIDq + cc]; }
      if (m1 > 0) { h1 = hsrc[(size_t)(m1 - 1) * HIDq + cc];
                    c1 = csrc[(size_t)(m1 - 1) * HIDq + cc]; }
      hv = 0.5f * (h0 + h1); cv = 0.5f * (c0 + c1);
    }
    sh_h[i] = (bf16_t)hv;
    sh_c[i] = cv;
  }
  __syncthreads();

  const bf16_t* xlane =
      feats + ((size_t)level * Nq + row0 + mlane) * FEAT + hh * 8;
  const bf16_t* hlane = sh_h + (size_t)mlane * HIDq + hh * 8;
  const bf16_t* wk = Wcat2 + (size_t)(wave * 64 + mlane) * 32 + hh * 8;

  v8f acc[16] = {};
  for (int kt = 0; kt < KT2; ++kt) {
    v16bf A = (kt < FEAT / 32) ? ldA(xlane + kt * 32)
                               : ldA(hlane + (kt - FEAT / 32) * 32);
    v16bf B0 = ldB(wk);
    v16bf B1 = ldB(wk + GSTRIDE);
    v16bf B2 = ldB(wk + 2 * GSTRIDE);
    v16bf B3 = ldB(wk + 3 * GSTRIDE);
#pragma unroll
    for (int jt = 0; jt < 4; ++jt) {
      v16bf N0, N1, N2, N3;
      if (jt < 3) {
        const bf16_t* q = wk + (jt + 1) * JSTRIDE;
        N0 = ldB(q);
        N1 = ldB(q + GSTRIDE);
        N2 = ldB(q + 2 * GSTRIDE);
        N3 = ldB(q + 3 * GSTRIDE);
      }
      acc[jt * 4 + 0] = wmma_bf16(A, B0, acc[jt * 4 + 0]);
      acc[jt * 4 + 1] = wmma_bf16(A, B1, acc[jt * 4 + 1]);
      acc[jt * 4 + 2] = wmma_bf16(A, B2, acc[jt * 4 + 2]);
      acc[jt * 4 + 3] = wmma_bf16(A, B3, acc[jt * 4 + 3]);
      if (jt < 3) { B0 = N0; B1 = N1; B2 = N2; B3 = N3; }
    }
    wk += (size_t)GQ * 32;
  }

#pragma unroll
  for (int jt = 0; jt < 4; ++jt) {
    int j = wave * 64 + jt * 16 + mlane;
    float bi = b_ih[j] + b_hh[j];
    float bfv = b_ih[HIDq + j] + b_hh[HIDq + j];
    float bg = b_ih[2 * HIDq + j] + b_hh[2 * HIDq + j];
    float bo = b_ih[3 * HIDq + j] + b_hh[3 * HIDq + j];
#pragma unroll
    for (int e = 0; e < 8; ++e) {
      int r = e + 8 * hh;
      float cin = sh_c[r * HIDq + j];
      float iv = sigmf(acc[jt * 4 + 0][e] + bi);
      float fv = sigmf(acc[jt * 4 + 1][e] + bfv);
      float gv = tanhf(acc[jt * 4 + 2][e] + bg);
      float ov = sigmf(acc[jt * 4 + 3][e] + bo);
      float cn = fv * cin + iv * gv;
      hdst[(size_t)(row0 + r) * HIDq + j] = ov * tanhf(cn);
      cdst[(size_t)(row0 + r) * HIDq + j] = cn;
    }
  }
}

// ---------------- heads (tiny FLOPs, scalar) ----------------
__global__ void bn512_kernel(const float* x, const float* g, const float* b,
                             float* z) {
  int col = blockIdx.x, tid = threadIdx.x;  // 128 threads, 512 rows
  float v[4], s = 0.f, q = 0.f;
#pragma unroll
  for (int i = 0; i < 4; ++i) {
    v[i] = x[(size_t)(tid + i * 128) * HIDq + col];
    s += v[i]; q += v[i] * v[i];
  }
  __shared__ float rs[128], rq[128];
  rs[tid] = s; rq[tid] = q; __syncthreads();
  for (int o = 64; o > 0; o >>= 1) {
    if (tid < o) { rs[tid] += rs[tid + o]; rq[tid] += rq[tid + o]; }
    __syncthreads();
  }
  float mean = rs[0] * (1.0f / Nq);
  float var = rq[0] * (1.0f / Nq) - mean * mean;
  float sc = g[col] * rsqrtf(var + EPSq), sh = b[col] - mean * sc;
#pragma unroll
  for (int i = 0; i < 4; ++i)
    z[(size_t)(tid + i * 128) * HIDq + col] = v[i] * sc + sh;
}

__global__ void t_mid_kernel(const float* z, const float* w2, const float* b2,
                             const float* g3, const float* b3, float* out) {
  int col = blockIdx.x, tid = threadIdx.x;  // 256 cols, 128 threads
  float v[4], s = 0.f, q = 0.f;
#pragma unroll
  for (int i = 0; i < 4; ++i) {
    int row = tid + i * 128;
    const float* zr = z + (size_t)row * HIDq;
    const float* wr = w2 + (size_t)col * HIDq;
    float acc = 0.f;
    for (int k = 0; k < HIDq; ++k) acc += zr[k] * wr[k];
    v[i] = fmaxf(acc + b2[col], 0.f);
    s += v[i]; q += v[i] * v[i];
  }
  __shared__ float rs[128], rq[128];
  rs[tid] = s; rq[tid] = q; __syncthreads();
  for (int o = 64; o > 0; o >>= 1) {
    if (tid < o) { rs[tid] += rs[tid + o]; rq[tid] += rq[tid + o]; }
    __syncthreads();
  }
  float mean = rs[0] * (1.0f / Nq);
  float var = rq[0] * (1.0f / Nq) - mean * mean;
  float sc = g3[col] * rsqrtf(var + EPSq), sh = b3[col] - mean * sc;
#pragma unroll
  for (int i = 0; i < 4; ++i)
    out[(size_t)(tid + i * 128) * MIDq + col] = v[i] * sc + sh;
}

__global__ void t_out_kernel(const float* tbn, const float* w3,
                             const float* b3, const float* ow, const float* ob,
                             float* out) {
  int row = blockIdx.x, c = threadIdx.x;  // 512 rows, 256 threads
  const float* tr = tbn + (size_t)row * MIDq;
  const float* wr = w3 + (size_t)c * MIDq;
  float s = 0.f;
  for (int k = 0; k < MIDq; ++k) s += tr[k] * wr[k];
  float u = fmaxf(s + b3[c], 0.f);
  __shared__ float red[256];
  red[c] = u * ow[c]; __syncthreads();
  for (int o = 128; o > 0; o >>= 1) {
    if (c < o) red[c] += red[c + o];
    __syncthreads();
  }
  if (c == 0) out[row] = 1.0f / (1.0f + __expf(-(red[0] + ob[0])));
}

// ---------------- host ----------------
extern "C" void kernel_launch(void* const* d_in, const int* in_sizes, int n_in,
                              void* d_out, int out_size, void* d_ws,
                              size_t ws_size, hipStream_t stream) {
  const float* operators = (const float*)d_in[0];
  const float* extra     = (const float*)d_in[1];
  const float* cond1     = (const float*)d_in[2];
  const float* samples   = (const float*)d_in[4];
  const float* cmask     = (const float*)d_in[5];
  const int*   mapping   = (const int*)d_in[6];
  const float* w_ih1 = (const float*)d_in[7];
  const float* w_hh1 = (const float*)d_in[8];
  const float* b_ih1 = (const float*)d_in[9];
  const float* b_hh1 = (const float*)d_in[10];
  const float* sample_w = (const float*)d_in[11];
  const float* sample_b = (const float*)d_in[12];
  const float* cond_w = (const float*)d_in[13];
  const float* cond_b = (const float*)d_in[14];
  const float* bn1_g = (const float*)d_in[15];
  const float* bn1_b = (const float*)d_in[16];
  const float* w_ih2 = (const float*)d_in[17];
  const float* w_hh2 = (const float*)d_in[18];
  const float* b_ih2 = (const float*)d_in[19];
  const float* b_hh2 = (const float*)d_in[20];
  const float* bn2_g = (const float*)d_in[21];
  const float* bn2_b = (const float*)d_in[22];
  const float* t1_w2 = (const float*)d_in[23];
  const float* t1_b2 = (const float*)d_in[24];
  const float* t2_w2 = (const float*)d_in[25];
  const float* t2_b2 = (const float*)d_in[26];
  const float* bn3_g = (const float*)d_in[27];
  const float* bn3_b = (const float*)d_in[28];
  const float* t1_w3 = (const float*)d_in[29];
  const float* t1_b3 = (const float*)d_in[30];
  const float* t2_w3 = (const float*)d_in[31];
  const float* t2_b3 = (const float*)d_in[32];
  const float* t1_ow = (const float*)d_in[33];
  const float* t1_ob = (const float*)d_in[34];
  const float* t2_ow = (const float*)d_in[35];
  const float* t2_ob = (const float*)d_in[36];
  float* out = (float*)d_out;

  uint8_t* base = (uint8_t*)d_ws;
  size_t off = 0;
  auto alloc = [&](size_t bytes) -> void* {
    void* p = base + off;
    off = (off + bytes + 255) & ~(size_t)255;
    return p;
  };
  bf16_t* Wcat1  = (bf16_t*)alloc((size_t)GQ * K1q * 2);
  bf16_t* Wcat2  = (bf16_t*)alloc((size_t)GQ * K2q * 2);
  bf16_t* swpad  = (bf16_t*)alloc((size_t)MIDq * 1024 * 2);
  bf16_t* cwbf   = (bf16_t*)alloc((size_t)MIDq * HIDq * 2);
  bf16_t* sampad = (bf16_t*)alloc((size_t)ROWS1 * 1024 * 2);
  bf16_t* h1     = (bf16_t*)alloc((size_t)ROWS1 * HIDq * 2);
  float*  condp  = (float*)alloc((size_t)ROWS1 * MIDq * 4);
  bf16_t* feats  = (bf16_t*)alloc((size_t)Lq * Nq * FEAT * 2);
  float*  hA     = (float*)alloc((size_t)Nq * HIDq * 4);
  float*  hB     = (float*)alloc((size_t)Nq * HIDq * 4);
  float*  cA     = (float*)alloc((size_t)Nq * HIDq * 4);
  float*  cB     = (float*)alloc((size_t)Nq * HIDq * 4);
  float*  z      = (float*)alloc((size_t)Nq * HIDq * 4);
  float*  t1bn   = (float*)alloc((size_t)Nq * MIDq * 4);
  float*  t2bn   = (float*)alloc((size_t)Nq * MIDq * 4);
  float*  stats  = (float*)alloc(1024 * 4);
  float*  sums = stats, * sqs = stats + 256;

  prep_weights_kernel<<<2048, 256, 0, stream>>>(w_ih1, w_hh1, w_ih2, w_hh2,
                                                sample_w, cond_w, Wcat1, Wcat2,
                                                swpad, cwbf, stats);
  prep_acts_kernel<<<4096, 256, 0, stream>>>(samples, operators, extra, sampad,
                                             feats);
  lstm1_kernel<<<ROWS1 / 32, 512, 0, stream>>>(cond1, Wcat1, b_ih1, b_hh1, h1);
  cond_gemm_kernel<<<ROWS1 / 16, 256, 0, stream>>>(h1, cwbf, cond_b, condp,
                                                   sums, sqs);
  bn1_apply_kernel<<<2048, 256, 0, stream>>>(condp, sums, sqs, bn1_g, bn1_b,
                                             feats);
  samp_gemm_kernel<<<ROWS1 / 16, 256, 0, stream>>>(sampad, swpad, sample_b,
                                                   cmask, feats);

  // tree LSTM: level L-1 first (zero state), then L-2..0 with gather
  lstm2_level_kernel<<<Nq / 16, 256, 0, stream>>>(feats, Wcat2, b_ih2, b_hh2,
                                                  mapping, hA, cA, hA, cA,
                                                  Lq - 1, 1);
  const float* hs = hA; const float* cs = cA;
  int flip = 0;
  for (int lvl = Lq - 2; lvl >= 0; --lvl) {
    float* hd = flip ? hA : hB;
    float* cd = flip ? cA : cB;
    lstm2_level_kernel<<<Nq / 16, 256, 0, stream>>>(feats, Wcat2, b_ih2, b_hh2,
                                                    mapping, hs, cs, hd, cd,
                                                    lvl, 0);
    hs = hd; cs = cd; flip ^= 1;
  }

  bn512_kernel<<<HIDq, 128, 0, stream>>>(hs, bn2_g, bn2_b, z);
  t_mid_kernel<<<MIDq, 128, 0, stream>>>(z, t1_w2, t1_b2, bn3_g, bn3_b, t1bn);
  t_mid_kernel<<<MIDq, 128, 0, stream>>>(z, t2_w2, t2_b2, bn3_g, bn3_b, t2bn);
  t_out_kernel<<<Nq, 256, 0, stream>>>(t1bn, t1_w3, t1_b3, t1_ow, t1_ob, out);
  t_out_kernel<<<Nq, 256, 0, stream>>>(t2bn, t2_w3, t2_b3, t2_ow, t2_ob,
                                       out + Nq);
}
